// GCN_Net_76776835383825
// MI455X (gfx1250) — matrix operand order repaired and checked by reading, not compiled, not run on hardware
//
#include <hip/hip_runtime.h>
#include <hip/hip_bf16.h>

// ---------------- CDNA5 WMMA types ----------------
typedef __attribute__((ext_vector_type(16))) __bf16 v16bf;
typedef __attribute__((ext_vector_type(8)))  float  v8f;
typedef __attribute__((ext_vector_type(8)))  int    v8i;

#define FEAT 128

__device__ __forceinline__ unsigned short f2bf(float f) {
  unsigned u = __builtin_bit_cast(unsigned, f);
  u += 0x7fffu + ((u >> 16) & 1u);           // round-to-nearest-even
  return (unsigned short)(u >> 16);
}
__device__ __forceinline__ float bf2f(unsigned short h) {
  unsigned u = ((unsigned)h) << 16;
  return __builtin_bit_cast(float, u);
}
__device__ __forceinline__ unsigned hash_u32(unsigned x) {
  x ^= x >> 16; x *= 0x7feb352du;
  x ^= x >> 15; x *= 0x846ca68bu;
  x ^= x >> 16; return x;
}

// ---------------- degree / normalization ----------------
__global__ void deg_init(float* deg, int n) {
  int i = blockIdx.x * blockDim.x + threadIdx.x;
  if (i < n) deg[i] = 1.0f;                  // self-loop contributes 1
}
__global__ void deg_count(const long long* __restrict__ dst, float* deg, int ne) {
  int e = blockIdx.x * blockDim.x + threadIdx.x;
  if (e < ne) atomicAdd(&deg[dst[e]], 1.0f);
}
__global__ void deg_finish(float* deg, int n) {
  int i = blockIdx.x * blockDim.x + threadIdx.x;
  if (i < n) deg[i] = rsqrtf(deg[i]);        // deg >= 1 always
}

// ---------------- fp32 -> bf16 conversions ----------------
__global__ void f32_to_bf16(const float* __restrict__ in, unsigned short* __restrict__ out, int total) {
  int i = blockIdx.x * blockDim.x + threadIdx.x;
  if (i < total) out[i] = f2bf(in[i]);
}
// Wt[n*128 + k] = bf16(W[k*128 + n])  (transpose so B fragments are k-contiguous)
__global__ void conv_w_t128(const float* __restrict__ W, unsigned short* __restrict__ Wt) {
  int idx = blockIdx.x * blockDim.x + threadIdx.x;   // 0 .. 16383
  int nn = idx & 127, kk = idx >> 7;
  Wt[(size_t)nn * 128 + kk] = f2bf(W[(size_t)kk * 128 + nn]);
}

// ---- WMMA GEMM + fused self-loop/bias epilogue -----------------------
// H[M,128]   = A[M,128] * B[128,128]                  (gathered by edge pass)
// AGG[M,128] = bias + dinv[m]^2 * H                    (scatter-add target)
// A row-major bf16; Bt n-major bf16 (Bt[n*128+k]); one wave per 16x16 tile.
__global__ void __launch_bounds__(256)
gemm_wmma_fused(const unsigned short* __restrict__ A,
                const unsigned short* __restrict__ Bt,
                const float* __restrict__ dinv,
                const float* __restrict__ bias,
                float* __restrict__ H,
                float* __restrict__ AGG) {
  const int lane = threadIdx.x & 31;
  const int wave = threadIdx.x >> 5;       // column tile 0..7
  const int half = lane >> 4;              // 0 or 1 (lane group)
  const int l16  = lane & 15;
  const int rowTile = blockIdx.x;

  const unsigned* Arow = (const unsigned*)(A  + ((size_t)rowTile * 16 + l16) * 128);
  const unsigned* Brow = (const unsigned*)(Bt + ((size_t)wave   * 16 + l16) * 128);

  v8f c = {0.f, 0.f, 0.f, 0.f, 0.f, 0.f, 0.f, 0.f};

#pragma unroll
  for (int ks = 0; ks < 4; ++ks) {         // K = 128 = 4 * 32
    v8i ai, bi;
    const int abase = ks * 16 + half * 8;  // dword index of kbase/2
#pragma unroll
    for (int i = 0; i < 4; ++i) ai[i]     = Arow[abase + i];       // K = kb+0..7
#pragma unroll
    for (int i = 0; i < 4; ++i) ai[4 + i] = Arow[abase + 4 + i];   // K = kb+8..15
    const int bbase = ks * 16 + half * 8;
#pragma unroll
    for (int i = 0; i < 8; ++i) bi[i]     = Brow[bbase + i];       // K contiguous

    v16bf a = __builtin_bit_cast(v16bf, ai);
    v16bf b = __builtin_bit_cast(v16bf, bi);
    c = __builtin_amdgcn_wmma_f32_16x16x32_bf16(false, a, false, b,
                                                (short)0, c, false, false);
  }

  // C layout: VGPR r -> M = m0 + r (m0 = rowTile*16 + 8*half), N = wave*16 + l16
  const int m0 = rowTile * 16 + 8 * half;
  const int n0 = wave * 16 + l16;
  const float bn = bias[n0];
  float*       Hb = H   + (size_t)m0 * 128 + n0;
  float*       Gb = AGG + (size_t)m0 * 128 + n0;
#pragma unroll
  for (int r = 0; r < 8; ++r) {
    float v  = c[r];
    float di = dinv[m0 + r];               // uniform across each 16-lane half
    Hb[(size_t)r * 128] = v;
    Gb[(size_t)r * 128] = bn + di * di * v;
  }
}

// ---------------- edge aggregation (scatter-add) ----------------
// one wave per edge, 128 features: float4 gather + 4 f32 atomics per lane
__global__ void __launch_bounds__(256)
agg_edges128(const long long* __restrict__ src, const long long* __restrict__ dst,
             const float* __restrict__ h, const float* __restrict__ dinv,
             float* __restrict__ out, int ne) {
  int e = (int)((blockIdx.x * (unsigned)blockDim.x + threadIdx.x) >> 5);
  if (e >= ne) return;
  int lane = threadIdx.x & 31;
  long long s = src[e], d = dst[e];
  float wgt = dinv[s] * dinv[d];
  float4 v = ((const float4*)(h + (size_t)s * 128))[lane];
  float* o = out + (size_t)d * 128 + lane * 4;
  atomicAdd(o + 0, v.x * wgt);
  atomicAdd(o + 1, v.y * wgt);
  atomicAdd(o + 2, v.z * wgt);
  atomicAdd(o + 3, v.w * wgt);
}

// 2-feature version (final layer): one thread per edge
__global__ void agg_edges2(const long long* __restrict__ src, const long long* __restrict__ dst,
                           const float* __restrict__ h, const float* __restrict__ dinv,
                           float* __restrict__ out, int ne) {
  int e = blockIdx.x * blockDim.x + threadIdx.x;
  if (e >= ne) return;
  long long s = src[e], d = dst[e];
  float wgt = dinv[s] * dinv[d];
  atomicAdd(&out[(size_t)d * 2 + 0], h[(size_t)s * 2 + 0] * wgt);
  atomicAdd(&out[(size_t)d * 2 + 1], h[(size_t)s * 2 + 1] * wgt);
}

// ---------------- leaky_relu + dropout(0.5) + bf16 repack ----------------
__global__ void act_drop(const float* __restrict__ in, unsigned short* __restrict__ out,
                         int total, unsigned seed) {
  int idx = blockIdx.x * blockDim.x + threadIdx.x;
  if (idx >= total) return;
  float v = in[idx];
  v = v > 0.f ? v : 0.01f * v;
  unsigned r = hash_u32(((unsigned)idx) * 2654435761u + seed);
  v = (r & 1u) ? (v * 2.0f) : 0.f;          // keep prob 0.5, scale 1/keep
  out[idx] = f2bf(v);
}

// ---------------- final 128 -> 2 dense layer (+ fused bias/self-loop) ----------------
__global__ void gemm_small_fused(const unsigned short* __restrict__ xb,
                                 const float* __restrict__ W3,
                                 const float* __restrict__ dinv,
                                 const float* __restrict__ b3,
                                 float* __restrict__ h3,
                                 float* __restrict__ agg3, int n) {
  int i = blockIdx.x * blockDim.x + threadIdx.x;
  if (i >= n) return;
  const unsigned* row = (const unsigned*)(xb + (size_t)i * 128);
  float c0 = 0.f, c1 = 0.f;
#pragma unroll
  for (int k2 = 0; k2 < 64; ++k2) {
    unsigned p = row[k2];
    float v0 = bf2f((unsigned short)(p & 0xffffu));
    float v1 = bf2f((unsigned short)(p >> 16));
    c0 += v0 * W3[4 * k2 + 0] + v1 * W3[4 * k2 + 2];
    c1 += v0 * W3[4 * k2 + 1] + v1 * W3[4 * k2 + 3];
  }
  h3[2 * i + 0] = c0;
  h3[2 * i + 1] = c1;
  float di2 = dinv[i] * dinv[i];
  agg3[2 * i + 0] = b3[0] + di2 * c0;
  agg3[2 * i + 1] = b3[1] + di2 * c1;
}

// ---------------- log_softmax over 2 classes ----------------
__global__ void log_softmax2(const float* __restrict__ in, float* __restrict__ out, int n) {
  int i = blockIdx.x * blockDim.x + threadIdx.x;
  if (i >= n) return;
  float a = in[2 * i], b = in[2 * i + 1];
  float m = fmaxf(a, b);
  float lse = m + __logf(__expf(a - m) + __expf(b - m));
  out[2 * i + 0] = a - lse;
  out[2 * i + 1] = b - lse;
}

// =====================================================================
static inline size_t align256(size_t x) { return (x + 255) & ~(size_t)255; }
#define CEIL(a, b) (((a) + (b) - 1) / (b))

extern "C" void kernel_launch(void* const* d_in, const int* in_sizes, int n_in,
                              void* d_out, int out_size, void* d_ws, size_t ws_size,
                              hipStream_t stream) {
  const float*     x   = (const float*)d_in[0];
  const long long* ei  = (const long long*)d_in[1];
  const float*     W1  = (const float*)d_in[2];
  const float*     b1  = (const float*)d_in[3];
  const float*     W2  = (const float*)d_in[4];
  const float*     b2  = (const float*)d_in[5];
  const float*     W3  = (const float*)d_in[6];
  const float*     b3  = (const float*)d_in[7];
  const int n  = in_sizes[0] / FEAT;     // 50000 nodes
  const int ne = in_sizes[1] / 2;        // 800000 edges
  const long long* src = ei;
  const long long* dst = ei + ne;

  // workspace carve-out
  char* w = (char*)d_ws;
  size_t off = 0;
  auto alloc = [&](size_t bytes) -> void* {
    void* p = w + off; off = align256(off + bytes); return p;
  };
  float*          dinv = (float*)alloc((size_t)n * 4);
  unsigned short* xb   = (unsigned short*)alloc((size_t)n * FEAT * 2);  // bf16 activations
  unsigned short* Wt   = (unsigned short*)alloc((size_t)FEAT * FEAT * 2);
  float*          h    = (float*)alloc((size_t)n * FEAT * 4);           // GEMM out (gathered)
  float*          agg  = (float*)alloc((size_t)n * FEAT * 4);           // scatter-add target
  float*          h3   = (float*)alloc((size_t)n * 2 * 4);
  float*          agg3 = (float*)alloc((size_t)n * 2 * 4);
  (void)ws_size; (void)n_in; (void)out_size;

  const int nf = n * FEAT;
  const dim3 B256(256);

  // ---- degree / dinv (shared by all three convs) ----
  deg_init  <<<CEIL(n, 256),  B256, 0, stream>>>(dinv, n);
  deg_count <<<CEIL(ne, 256), B256, 0, stream>>>(dst, dinv, ne);
  deg_finish<<<CEIL(n, 256),  B256, 0, stream>>>(dinv, n);

  // ---- layer 1 ----
  f32_to_bf16<<<CEIL(nf, 256), B256, 0, stream>>>(x, xb, nf);
  conv_w_t128<<<64, B256, 0, stream>>>(W1, Wt);
  gemm_wmma_fused<<<n / 16, B256, 0, stream>>>(xb, Wt, dinv, b1, h, agg);
  agg_edges128<<<CEIL((size_t)ne * 32, 256), B256, 0, stream>>>(src, dst, h, dinv, agg, ne);
  act_drop    <<<CEIL(nf, 256), B256, 0, stream>>>(agg, xb, nf, 0x9E3779B9u);

  // ---- layer 2 ----
  conv_w_t128<<<64, B256, 0, stream>>>(W2, Wt);
  gemm_wmma_fused<<<n / 16, B256, 0, stream>>>(xb, Wt, dinv, b2, h, agg);
  agg_edges128<<<CEIL((size_t)ne * 32, 256), B256, 0, stream>>>(src, dst, h, dinv, agg, ne);
  act_drop    <<<CEIL(nf, 256), B256, 0, stream>>>(agg, xb, nf, 0x85EBCA6Bu);

  // ---- layer 3 (128 -> 2) ----
  gemm_small_fused<<<CEIL(n, 256), B256, 0, stream>>>(xb, W3, dinv, b3, h3, agg3, n);
  agg_edges2<<<CEIL(ne, 256), B256, 0, stream>>>(src, dst, h3, dinv, agg3, ne);

  // ---- log_softmax -> d_out ----
  log_softmax2<<<CEIL(n, 256), B256, 0, stream>>>(agg3, (float*)d_out, n);
}
